// PatchTransformerNepaPartSeg_12884901888632
// MI455X (gfx1250) — matrix operand order repaired and checked by reading, not compiled, not run on hardware
//
#include <hip/hip_runtime.h>
#include <hip/hip_bf16.h>
#include <math.h>

#define BB 16
#define NN 2048
#define GG 128
#define KK 32
#define DD 384
#define LL 12
#define HH 6
#define DHH 64
#define DFFF 1536
#define SS 130
#define NPP 50
#define NCC 16
#define LDD 64
#define INDIM 835
#define INP 864   // IN_DIM padded to multiple of 32 for WMMA K

typedef _Float16 f16;
typedef __attribute__((ext_vector_type(16))) _Float16 v16h;
typedef __attribute__((ext_vector_type(8)))  _Float16 v8h;
typedef __attribute__((ext_vector_type(8)))  float    v8f;

static inline int npad256(int n) { return ((n + 255) / 256) * 256; }

// ---------------------------------------------------------------------------
// Generic WMMA GEMM: C[M,N] = act( A16[M,Kp] @ Wt16[Npad,Kp]^T + bias [bn] ) + resid
// A row-major f16, Wt row-major f16 (B pre-transposed, N zero-padded to the
// grid's full column coverage so the inner loop has NO guards). Kp % 32 == 0,
// M % 32 == 0. Per wave: 16x64 tile (4 accumulators, A fragment reused).
// ---------------------------------------------------------------------------
__global__ __launch_bounds__(256) void gemm_wmma_kernel(
    const f16* __restrict__ A, const f16* __restrict__ Wt,
    const float* __restrict__ bias, const float* __restrict__ bng,
    const float* __restrict__ bnb, const float* __restrict__ resid,
    float* __restrict__ Cf, f16* __restrict__ Ch,
    int M, int N, int Kp, int act)
{
  const int lane = threadIdx.x & 31;
  const int wave = threadIdx.x >> 5;
  const int m0 = blockIdx.y * 32 + (wave >> 2) * 16;
  const int n0 = blockIdx.x * 256 + (wave & 3) * 64;
  const int lr = lane & 15;   // row (A) / col (B,C) within 16
  const int hi = lane >> 4;   // half-wave select

  v8f acc[4] = {};
  const f16* arow = A + (size_t)(m0 + lr) * Kp + (hi << 4);
  const f16* brow0 = Wt + (size_t)(n0 + 0 * 16 + lr) * Kp + (hi << 4);
  const f16* brow1 = Wt + (size_t)(n0 + 1 * 16 + lr) * Kp + (hi << 4);
  const f16* brow2 = Wt + (size_t)(n0 + 2 * 16 + lr) * Kp + (hi << 4);
  const f16* brow3 = Wt + (size_t)(n0 + 3 * 16 + lr) * Kp + (hi << 4);

  for (int kk = 0; kk < Kp; kk += 32) {
    __builtin_prefetch(arow + kk + 128, 0, 1);
    v8h a0 = *(const v8h*)(arow + kk);
    v8h a1 = *(const v8h*)(arow + kk + 8);
    v16h af;
#pragma unroll
    for (int i = 0; i < 8; ++i) { af[i] = a0[i]; af[i + 8] = a1[i]; }

    const f16* bp[4] = { brow0 + kk, brow1 + kk, brow2 + kk, brow3 + kk };
#pragma unroll
    for (int sub = 0; sub < 4; ++sub) {
      v8h b0 = *(const v8h*)(bp[sub]);
      v8h b1 = *(const v8h*)(bp[sub] + 8);
      v16h bf;
#pragma unroll
      for (int i = 0; i < 8; ++i) { bf[i] = b0[i]; bf[i + 8] = b1[i]; }
      acc[sub] = __builtin_amdgcn_wmma_f32_16x16x32_f16(
          false, af, false, bf, (short)0, acc[sub], false, false);
    }
  }

  const float bninv = rsqrtf(1.0f + 1e-5f);
#pragma unroll
  for (int sub = 0; sub < 4; ++sub) {
    int n = n0 + sub * 16 + lr;
    if (n >= N) continue;
#pragma unroll
    for (int r = 0; r < 8; ++r) {
      int m = m0 + r + (hi << 3);
      float v = acc[sub][r];
      if (bias) v += bias[n];
      if (bng)  v = bng[n] * v * bninv + bnb[n];
      if (act == 1) {
        v = v > 0.f ? v : 0.f;
      } else if (act == 2) {
        float x3 = v * v * v;
        v = 0.5f * v * (1.f + tanhf(0.7978845608028654f * (v + 0.044715f * x3)));
      }
      size_t o = (size_t)m * N + n;
      if (resid) v += resid[o];
      if (Cf) Cf[o] = v;
      if (Ch) Ch[o] = (f16)v;
    }
  }
}

// ---------------------------------------------------------------------------
// Weight transpose + f32->f16 convert with K and N zero-padding:
// W[K,N] -> Wt[Npad,Kpad]
// ---------------------------------------------------------------------------
__global__ void wtconv_kernel(const float* __restrict__ W, f16* __restrict__ Wt,
                              int Kdim, int Ndim, int Kpad, int Npad)
{
  size_t total = (size_t)Npad * Kpad;
  for (size_t i = (size_t)blockIdx.x * blockDim.x + threadIdx.x; i < total;
       i += (size_t)gridDim.x * blockDim.x) {
    int k = (int)(i % Kpad);
    int n = (int)(i / Kpad);
    Wt[i] = (k < Kdim && n < Ndim) ? (f16)W[(size_t)k * Ndim + n] : (f16)0.f;
  }
}

// ---------------------------------------------------------------------------
// Centers + 32-NN selection (matches lax.top_k tie-break: lower index first)
// ---------------------------------------------------------------------------
__global__ __launch_bounds__(256) void knn_kernel(const float* __restrict__ xyz,
    float* __restrict__ centers, int* __restrict__ nidx)
{
  const int bg = blockIdx.x;
  const int b = bg / GG, g = bg % GG;
  const int t = threadIdx.x;
  const float* pb = xyz + (size_t)b * NN * 3;
  float cx = pb[(g * 16) * 3 + 0];
  float cy = pb[(g * 16) * 3 + 1];
  float cz = pb[(g * 16) * 3 + 2];
  if (t == 0) {
    centers[bg * 3 + 0] = cx; centers[bg * 3 + 1] = cy; centers[bg * 3 + 2] = cz;
  }
  __shared__ float sd[NN];
  for (int i = t; i < NN; i += 256) {
    float dx = pb[i * 3] - cx, dy = pb[i * 3 + 1] - cy, dz = pb[i * 3 + 2] - cz;
    sd[i] = dx * dx + dy * dy + dz * dz;
  }
  __syncthreads();
  __shared__ float rv[256];
  __shared__ int   ri[256];
  for (int sel = 0; sel < KK; ++sel) {
    float bv = 3.4e38f; int bi = NN;
    for (int i = t; i < NN; i += 256) {
      float v = sd[i];
      if (v < bv) { bv = v; bi = i; }
    }
    rv[t] = bv; ri[t] = bi;
    __syncthreads();
    for (int off = 128; off > 0; off >>= 1) {
      if (t < off) {
        float v2 = rv[t + off]; int i2 = ri[t + off];
        if (v2 < rv[t] || (v2 == rv[t] && i2 < ri[t])) { rv[t] = v2; ri[t] = i2; }
      }
      __syncthreads();
    }
    if (t == 0) {
      nidx[(size_t)bg * KK + sel] = ri[0];
      sd[ri[0]] = 3.4e38f;
    }
    __syncthreads();
  }
}

// ---------------------------------------------------------------------------
// Mini-PointNet per patch: relu(rel@We1+be1) -> max over K -> @We2+be2
// ---------------------------------------------------------------------------
__global__ __launch_bounds__(128) void pointnet_kernel(const float* __restrict__ xyz,
    const float* __restrict__ centers, const int* __restrict__ nidx,
    const float* __restrict__ We1, const float* __restrict__ be1,
    const float* __restrict__ We2, const float* __restrict__ be2,
    float* __restrict__ tokens)
{
  int bg = blockIdx.x, b = bg / GG, t = threadIdx.x;
  const float* pb = xyz + (size_t)b * NN * 3;
  float cx = centers[bg * 3], cy = centers[bg * 3 + 1], cz = centers[bg * 3 + 2];
  float w0 = We1[t], w1 = We1[128 + t], w2 = We1[256 + t], b0 = be1[t];
  float mx = -3.4e38f;
  for (int k = 0; k < KK; ++k) {
    int pid = nidx[(size_t)bg * KK + k];
    float rx = pb[pid * 3] - cx, ry = pb[pid * 3 + 1] - cy, rz = pb[pid * 3 + 2] - cz;
    float hv = rx * w0 + ry * w1 + rz * w2 + b0;
    hv = hv > 0.f ? hv : 0.f;
    mx = fmaxf(mx, hv);
  }
  __shared__ float m[128];
  m[t] = mx;
  __syncthreads();
  for (int o = t; o < DD; o += 128) {
    float s = be2[o];
    for (int i = 0; i < 128; ++i) s += m[i] * We2[i * DD + o];
    tokens[(size_t)bg * DD + o] = s;
  }
}

// ---------------------------------------------------------------------------
// Sequence assembly: bos/token/eos + type_emb + pos-MLP(center)
// ---------------------------------------------------------------------------
__global__ __launch_bounds__(128) void embed_kernel(const float* __restrict__ tokens,
    const float* __restrict__ centers, const float* __restrict__ bos,
    const float* __restrict__ eos, const float* __restrict__ type_emb,
    const float* __restrict__ Wp1, const float* __restrict__ bp1,
    const float* __restrict__ Wp2, const float* __restrict__ bp2,
    float* __restrict__ x)
{
  int bs = blockIdx.x, b = bs / SS, s = bs % SS, t = threadIdx.x;
  float c0 = 0.f, c1 = 0.f, c2 = 0.f; int ty = 0;
  if (s >= 1 && s <= GG) {
    int g = s - 1;
    c0 = centers[(b * GG + g) * 3];
    c1 = centers[(b * GG + g) * 3 + 1];
    c2 = centers[(b * GG + g) * 3 + 2];
    ty = 1;
  } else if (s == SS - 1) {
    ty = 2;
  }
  __shared__ float ph[128];
  float hv = c0 * Wp1[t] + c1 * Wp1[128 + t] + c2 * Wp1[256 + t] + bp1[t];
  ph[t] = hv > 0.f ? hv : 0.f;
  __syncthreads();
  for (int d = t; d < DD; d += 128) {
    float p = bp2[d];
    for (int i = 0; i < 128; ++i) p += ph[i] * Wp2[i * DD + d];
    float basev = (ty == 0) ? bos[d] : (ty == 2) ? eos[d]
                  : tokens[((size_t)b * GG + (s - 1)) * DD + d];
    x[(size_t)bs * DD + d] = basev + type_emb[ty * DD + d] + p;
  }
}

// ---------------------------------------------------------------------------
// LayerNorm over D=384, writes f16 (GEMM input) and/or f32
// ---------------------------------------------------------------------------
__global__ __launch_bounds__(128) void ln_kernel(const float* __restrict__ x,
    const float* __restrict__ g, const float* __restrict__ b,
    f16* __restrict__ o16, float* __restrict__ o32)
{
  int row = blockIdx.x, t = threadIdx.x;
  const float* xr = x + (size_t)row * DD;
  float v0 = xr[t], v1 = xr[t + 128], v2 = xr[t + 256];
  __shared__ float s1[128], s2[128];
  s1[t] = v0 + v1 + v2;
  s2[t] = v0 * v0 + v1 * v1 + v2 * v2;
  __syncthreads();
  for (int off = 64; off > 0; off >>= 1) {
    if (t < off) { s1[t] += s1[t + off]; s2[t] += s2[t + off]; }
    __syncthreads();
  }
  float mean = s1[0] * (1.f / DD);
  float var  = s2[0] * (1.f / DD) - mean * mean;
  float rstd = rsqrtf(var + 1e-5f);
#pragma unroll
  for (int j = 0; j < 3; ++j) {
    int d = t + j * 128;
    float vv = (j == 0) ? v0 : (j == 1) ? v1 : v2;
    float o = g[d] * (vv - mean) * rstd + b[d];
    if (o16) o16[(size_t)row * DD + d] = (f16)o;
    if (o32) o32[(size_t)row * DD + d] = o;
  }
}

// ---------------------------------------------------------------------------
// Fused attention for one (b, h, q): scores, softmax, att@V -> f16 output
// qkv layout per row: [q(384) | k(384) | v(384)]
// ---------------------------------------------------------------------------
__global__ __launch_bounds__(128) void attn_kernel(const float* __restrict__ qkv,
                                                   f16* __restrict__ out16)
{
  int blk = blockIdx.x;
  int q = blk % SS, h = (blk / SS) % HH, b = blk / (SS * HH);
  int t = threadIdx.x;
  const float* base = qkv + (size_t)b * SS * (3 * DD);
  __shared__ float qv[DHH];
  __shared__ float sc[SS];
  __shared__ float red[128];
  if (t < DHH) qv[t] = base[(size_t)q * (3 * DD) + h * DHH + t];
  __syncthreads();
  for (int k = t; k < SS; k += 128) {
    const float* kr = base + (size_t)k * (3 * DD) + DD + h * DHH;
    float s = 0.f;
#pragma unroll 8
    for (int d = 0; d < DHH; ++d) s += qv[d] * kr[d];
    sc[k] = s * 0.125f;   // 1/sqrt(64)
  }
  __syncthreads();
  float lm = -3.4e38f;
  for (int k = t; k < SS; k += 128) lm = fmaxf(lm, sc[k]);
  red[t] = lm;
  __syncthreads();
  for (int off = 64; off > 0; off >>= 1) {
    if (t < off) red[t] = fmaxf(red[t], red[t + off]);
    __syncthreads();
  }
  float mx = red[0];
  __syncthreads();
  float ls = 0.f;
  for (int k = t; k < SS; k += 128) { float e = __expf(sc[k] - mx); sc[k] = e; ls += e; }
  red[t] = ls;
  __syncthreads();
  for (int off = 64; off > 0; off >>= 1) {
    if (t < off) red[t] += red[t + off];
    __syncthreads();
  }
  float inv = 1.f / red[0];
  if (t < DHH) {
    float o = 0.f;
    for (int k = 0; k < SS; ++k)
      o += sc[k] * base[(size_t)k * (3 * DD) + 2 * DD + h * DHH + t];
    out16[((size_t)b * SS + q) * DD + h * DHH + t] = (f16)(o * inv);
  }
}

// ---------------------------------------------------------------------------
// Nearest-center assignment per point (argmin, first-min tie-break)
// ---------------------------------------------------------------------------
__global__ void assign_kernel(const float* __restrict__ xyz,
                              const float* __restrict__ centers,
                              int* __restrict__ assign)
{
  int i = blockIdx.x * blockDim.x + threadIdx.x;
  if (i >= BB * NN) return;
  int b = i / NN;
  float px = xyz[(size_t)i * 3], py = xyz[(size_t)i * 3 + 1], pz = xyz[(size_t)i * 3 + 2];
  const float* cb = centers + b * GG * 3;
  float bv = 3.4e38f; int bg = 0;
  for (int g = 0; g < GG; ++g) {
    float dx = px - cb[g * 3], dy = py - cb[g * 3 + 1], dz = pz - cb[g * 3 + 2];
    float d = dx * dx + dy * dy + dz * dz;
    if (d < bv) { bv = d; bg = g; }
  }
  assign[i] = bg;
}

__global__ __launch_bounds__(128) void gmax_kernel(const float* __restrict__ hfin,
                                                   float* __restrict__ gf)
{
  int b = blockIdx.x, t = threadIdx.x;
  for (int d = t; d < DD; d += 128) {
    float m = -3.4e38f;
    for (int g = 0; g < GG; ++g)
      m = fmaxf(m, hfin[((size_t)b * SS + 1 + g) * DD + d]);
    gf[b * DD + d] = m;
  }
}

__global__ void cls_kernel(const int* __restrict__ lbl, const float* __restrict__ Wl,
                           const float* __restrict__ bl, float* __restrict__ clsf)
{
  int b = blockIdx.x, t = threadIdx.x;   // 64 threads
  float v = Wl[lbl[b] * LDD + t] + bl[t];
  clsf[b * LDD + t] = v > 0.f ? v : 0.f;
}

// feat16[row, c]: [point_feat(384) | global(384) | cls(64) | xyz(3) | pad(29)]
__global__ void feat_kernel(const float* __restrict__ hfin, const float* __restrict__ gf,
    const float* __restrict__ clsf, const float* __restrict__ xyz,
    const int* __restrict__ assign, f16* __restrict__ feat)
{
  size_t total = (size_t)BB * NN * INP;
  for (size_t idx = (size_t)blockIdx.x * blockDim.x + threadIdx.x; idx < total;
       idx += (size_t)gridDim.x * blockDim.x) {
    int c = (int)(idx % INP);
    size_t row = idx / INP;
    int b = (int)(row / NN);
    float v;
    if (c < DD)                 v = hfin[((size_t)b * SS + 1 + assign[row]) * DD + c];
    else if (c < 2 * DD)        v = gf[b * DD + (c - DD)];
    else if (c < 2 * DD + LDD)  v = clsf[b * LDD + (c - 2 * DD)];
    else if (c < INDIM)         v = xyz[row * 3 + (c - 2 * DD - LDD)];
    else                        v = 0.f;
    feat[idx] = (f16)v;
  }
}

// ---------------------------------------------------------------------------
extern "C" void kernel_launch(void* const* d_in, const int* in_sizes, int n_in,
                              void* d_out, int out_size, void* d_ws, size_t ws_size,
                              hipStream_t stream) {
  (void)in_sizes; (void)n_in; (void)out_size;
  const float* xyz      = (const float*)d_in[0];
  const int*   cls_lbl  = (const int*)d_in[1];
  const float* We1 = (const float*)d_in[2];  const float* be1 = (const float*)d_in[3];
  const float* We2 = (const float*)d_in[4];  const float* be2 = (const float*)d_in[5];
  const float* Wp1 = (const float*)d_in[6];  const float* bp1 = (const float*)d_in[7];
  const float* Wp2 = (const float*)d_in[8];  const float* bp2 = (const float*)d_in[9];
  const float* bos = (const float*)d_in[10]; const float* eos = (const float*)d_in[11];
  const float* type_emb = (const float*)d_in[12];
  const float* ln1_g = (const float*)d_in[13]; const float* ln1_b = (const float*)d_in[14];
  const float* Wqkv  = (const float*)d_in[15]; const float* bqkv  = (const float*)d_in[16];
  const float* Wo    = (const float*)d_in[17]; const float* bo    = (const float*)d_in[18];
  const float* ln2_g = (const float*)d_in[19]; const float* ln2_b = (const float*)d_in[20];
  const float* Wf1   = (const float*)d_in[21]; const float* bf1   = (const float*)d_in[22];
  const float* Wf2   = (const float*)d_in[23]; const float* bf2   = (const float*)d_in[24];
  const float* lnf_g = (const float*)d_in[25]; const float* lnf_b = (const float*)d_in[26];
  const float* Wl    = (const float*)d_in[27]; const float* bl    = (const float*)d_in[28];
  const float* Wc1   = (const float*)d_in[29]; const float* bc1   = (const float*)d_in[30];
  const float* bn1_g = (const float*)d_in[31]; const float* bn1_b = (const float*)d_in[32];
  const float* Wc2   = (const float*)d_in[33]; const float* bc2   = (const float*)d_in[34];
  const float* bn2_g = (const float*)d_in[35]; const float* bn2_b = (const float*)d_in[36];
  const float* Wc3   = (const float*)d_in[37]; const float* bc3   = (const float*)d_in[38];

  // column-padded weight row counts (grid coverage of each GEMM's N)
  const int NP_QKV = npad256(3 * DD);   // 1280
  const int NP_D   = npad256(DD);       // 512
  const int NP_F1  = npad256(DFFF);     // 1536
  const int NP_C1  = npad256(512);      // 512
  const int NP_C2  = npad256(256);      // 256
  const int NP_C3  = npad256(NPP);      // 256

  // ---- workspace layout ----
  size_t off = 0;
  char* base = (char*)d_ws;
  auto alloc = [&](size_t bytes) -> char* {
    char* p = base + off;
    off += (bytes + 255) & ~(size_t)255;
    return p;
  };
  float* centers = (float*)alloc((size_t)BB * GG * 3 * 4);
  int*   nidx    = (int*)  alloc((size_t)BB * GG * KK * 4);
  float* tokens  = (float*)alloc((size_t)BB * GG * DD * 4);
  float* x       = (float*)alloc((size_t)BB * SS * DD * 4);
  f16*   xln16   = (f16*)  alloc((size_t)BB * SS * DD * 2);
  float* qkv     = (float*)alloc((size_t)BB * SS * 3 * DD * 4);
  f16*   attno16 = (f16*)  alloc((size_t)BB * SS * DD * 2);
  f16*   h16     = (f16*)  alloc((size_t)BB * SS * DFFF * 2);
  float* hfin    = (float*)alloc((size_t)BB * SS * DD * 4);
  f16*   wt_qkv  = (f16*)  alloc((size_t)LL * NP_QKV * DD * 2);
  f16*   wt_o    = (f16*)  alloc((size_t)LL * NP_D * DD * 2);
  f16*   wt_f1   = (f16*)  alloc((size_t)LL * NP_F1 * DD * 2);
  f16*   wt_f2   = (f16*)  alloc((size_t)LL * NP_D * DFFF * 2);
  f16*   wt_c1   = (f16*)  alloc((size_t)NP_C1 * INP * 2);
  f16*   wt_c2   = (f16*)  alloc((size_t)NP_C2 * 512 * 2);
  f16*   wt_c3   = (f16*)  alloc((size_t)NP_C3 * 256 * 2);
  int*   assign  = (int*)  alloc((size_t)BB * NN * 4);
  float* gf      = (float*)alloc((size_t)BB * DD * 4);
  float* clsf    = (float*)alloc((size_t)BB * LDD * 4);
  f16*   feat16  = (f16*)  alloc((size_t)BB * NN * INP * 2);
  f16*   y1_16   = (f16*)  alloc((size_t)BB * NN * 512 * 2);
  f16*   y2_16   = (f16*)  alloc((size_t)BB * NN * 256 * 2);
  if (off > ws_size) return;   // workspace too small; cannot proceed safely

  const int M = BB * SS;       // 2080, multiple of 32
  const int MP = BB * NN;      // 32768

  auto gemm = [&](const f16* A, const f16* Wt, const float* bias,
                  const float* bng, const float* bnb, const float* resid,
                  float* Cf, f16* Ch, int Mm, int Nn, int Kp, int act) {
    dim3 grid((Nn + 255) / 256, Mm / 32);
    gemm_wmma_kernel<<<grid, 256, 0, stream>>>(A, Wt, bias, bng, bnb, resid,
                                               Cf, Ch, Mm, Nn, Kp, act);
  };
  auto wconv = [&](const float* W, f16* Wt, int Kd, int Nd, int Kpad, int Npad) {
    size_t total = (size_t)Npad * Kpad;
    int blocks = (int)((total + 255) / 256);
    if (blocks > 4096) blocks = 4096;
    wtconv_kernel<<<blocks, 256, 0, stream>>>(W, Wt, Kd, Nd, Kpad, Npad);
  };

  // ---- convert all weights to transposed f16 (WMMA B operand layout) ----
  for (int l = 0; l < LL; ++l) {
    wconv(Wqkv + (size_t)l * DD * 3 * DD, wt_qkv + (size_t)l * NP_QKV * DD, DD, 3 * DD, DD, NP_QKV);
    wconv(Wo   + (size_t)l * DD * DD,     wt_o   + (size_t)l * NP_D * DD,   DD, DD,     DD, NP_D);
    wconv(Wf1  + (size_t)l * DD * DFFF,   wt_f1  + (size_t)l * NP_F1 * DD,  DD, DFFF,   DD, NP_F1);
    wconv(Wf2  + (size_t)l * DFFF * DD,   wt_f2  + (size_t)l * NP_D * DFFF, DFFF, DD,   DFFF, NP_D);
  }
  wconv(Wc1, wt_c1, INDIM, 512, INP, NP_C1);
  wconv(Wc2, wt_c2, 512, 256, 512, NP_C2);
  wconv(Wc3, wt_c3, 256, NPP, 256, NP_C3);

  // ---- tokenization ----
  knn_kernel<<<BB * GG, 256, 0, stream>>>(xyz, centers, nidx);
  pointnet_kernel<<<BB * GG, 128, 0, stream>>>(xyz, centers, nidx, We1, be1, We2, be2, tokens);
  embed_kernel<<<M, 128, 0, stream>>>(tokens, centers, bos, eos, type_emb,
                                      Wp1, bp1, Wp2, bp2, x);

  // ---- transformer ----
  for (int l = 0; l < LL; ++l) {
    ln_kernel<<<M, 128, 0, stream>>>(x, ln1_g + l * DD, ln1_b + l * DD, xln16, nullptr);
    gemm(xln16, wt_qkv + (size_t)l * NP_QKV * DD, bqkv + (size_t)l * 3 * DD,
         nullptr, nullptr, nullptr, qkv, nullptr, M, 3 * DD, DD, 0);
    attn_kernel<<<BB * HH * SS, 128, 0, stream>>>(qkv, attno16);
    gemm(attno16, wt_o + (size_t)l * NP_D * DD, bo + (size_t)l * DD,
         nullptr, nullptr, x, x, nullptr, M, DD, DD, 0);
    ln_kernel<<<M, 128, 0, stream>>>(x, ln2_g + l * DD, ln2_b + l * DD, xln16, nullptr);
    gemm(xln16, wt_f1 + (size_t)l * NP_F1 * DD, bf1 + (size_t)l * DFFF,
         nullptr, nullptr, nullptr, nullptr, h16, M, DFFF, DD, 2 /*gelu*/);
    gemm(h16, wt_f2 + (size_t)l * NP_D * DFFF, bf2 + (size_t)l * DD,
         nullptr, nullptr, x, x, nullptr, M, DD, DFFF, 0);
  }
  ln_kernel<<<M, 128, 0, stream>>>(x, lnf_g, lnf_b, nullptr, hfin);

  // ---- per-point features ----
  assign_kernel<<<(BB * NN + 255) / 256, 256, 0, stream>>>(xyz, centers, assign);
  gmax_kernel<<<BB, 128, 0, stream>>>(hfin, gf);
  cls_kernel<<<BB, LDD, 0, stream>>>(cls_lbl, Wl, bl, clsf);
  feat_kernel<<<8192, 256, 0, stream>>>(hfin, gf, clsf, xyz, assign, feat16);

  // ---- classifier head (bn folded into GEMM epilogue) ----
  gemm(feat16, wt_c1, bc1, bn1_g, bn1_b, nullptr, nullptr, y1_16, MP, 512, INP, 1);
  gemm(y1_16,  wt_c2, bc2, bn2_g, bn2_b, nullptr, nullptr, y2_16, MP, 256, 512, 1);
  gemm(y2_16,  wt_c3, bc3, nullptr, nullptr, nullptr, (float*)d_out, nullptr, MP, NPP, 256, 0);
}
